// Convolve_83966610636967
// MI455X (gfx1250) — matrix operand order repaired and checked by reading, not compiled, not run on hardware
//
#include <hip/hip_runtime.h>

namespace {

constexpr int Sn   = 128;
constexpr int Nn   = 256;
constexpr int CIN  = 128;
constexpr int COUT = 128;
constexpr int R    = 4;
constexpr int R1   = 5;
constexpr float EPSF = 1e-7f;

typedef __attribute__((ext_vector_type(16))) __bf16 v16bf;
typedef __attribute__((ext_vector_type(8)))  __bf16 v8bf;
typedef __attribute__((ext_vector_type(8)))  float  v8f;
typedef __attribute__((ext_vector_type(4)))  float  v4f;

// ---- LDS carve (bytes) ----
constexpr unsigned XROW_OFF = 0;                        // bf16 [256][128]   x rows (full s)
constexpr unsigned WT_OFF   = XROW_OFF + 256*128*2;     // bf16 [2][128][128] W^T per r-pair
constexpr unsigned PT_OFF   = WT_OFF   + 2*128*128*2;   // bf16 [2][128][256] P^T[b][j] per r-pair
constexpr unsigned ATL_OFF  = PT_OFF   + 2*128*256*2;   // bf16 [8 waves][16][32] norm-scaled A tiles
constexpr unsigned ARAW_OFF = ATL_OFF  + 8*16*32*2;     // f32  [8 waves][16][32][2] raw async A tiles
constexpr unsigned NRM_OFF  = ARAW_OFF + 8*16*32*2*4;   // f32  [128][4]  1/(rowsum+eps)
constexpr unsigned LDS_BYTES= NRM_OFF  + 128*4*4;       // = 305,152 B (< 320 KB WGP LDS)

#if defined(__HIP_DEVICE_COMPILE__)
typedef __attribute__((address_space(3))) int  as3_int;
#endif

// Async global->LDS raw copy (ASYNCcnt-tracked), builtin if declared else raw ISA.
// ROCm prototype (probe-derived): (int* global_src_flat, as3 int* lds_dst, imm offset, imm cpol)
__device__ __forceinline__ void async_copy_b32(const void* g, void* l) {
#if defined(__HIP_DEVICE_COMPILE__)
# if __has_builtin(__builtin_amdgcn_global_load_async_to_lds_b32)
  __builtin_amdgcn_global_load_async_to_lds_b32((int*)g, (as3_int*)l, 0, 0);
# else
  unsigned lofs = (unsigned)(unsigned long long)l;   // 32-bit LDS offset
  asm volatile("global_load_async_to_lds_b32 %0, %1, off"
               :: "v"(lofs), "v"(g) : "memory");
# endif
#else
  (void)g; (void)l;
#endif
}

__device__ __forceinline__ void wait_async0() {
#if defined(__HIP_DEVICE_COMPILE__)
# if __has_builtin(__builtin_amdgcn_s_wait_asynccnt)
  __builtin_amdgcn_s_wait_asynccnt(0);
# else
  asm volatile("s_wait_asynccnt 0x0" ::: "memory");
# endif
#endif
}

__device__ __forceinline__ __bf16 f2bf(float f) {
  unsigned u = __builtin_bit_cast(unsigned, f);
  u += 0x7fffu + ((u >> 16) & 1u);                      // round-to-nearest-even
  unsigned short h = (unsigned short)(u >> 16);
  return __builtin_bit_cast(__bf16, h);
}

// Build a v16bf A/B fragment from two 16-byte LDS chunks (ds_load_b128 x2).
__device__ __forceinline__ v16bf frag(const __bf16* p0, const __bf16* p1) {
  v8bf lo = *(const v8bf*)p0;
  v8bf hi = *(const v8bf*)p1;
  return __builtin_shufflevector(lo, hi, 0,1,2,3,4,5,6,7,8,9,10,11,12,13,14,15);
}

__device__ __forceinline__ v8f wmma_bf16(v16bf a, v16bf b, v8f c) {
  // (neg_a, A, neg_b, B, c_mod, C, reuse_a, reuse_b)
  return __builtin_amdgcn_wmma_f32_16x16x32_bf16(false, a, false, b, (short)0, c, false, false);
}

__global__ __launch_bounds__(256, 1)
void fused_attnconv(const float* __restrict__ A, const float* __restrict__ X,
                    const float* __restrict__ W, const float* __restrict__ LW,
                    const float* __restrict__ LB, float* __restrict__ OUT)
{
  extern __shared__ char smem[];
  __bf16* xrow = (__bf16*)(smem + XROW_OFF);
  __bf16* wt_  = (__bf16*)(smem + WT_OFF);
  __bf16* pt_  = (__bf16*)(smem + PT_OFF);
  __bf16* atl  = (__bf16*)(smem + ATL_OFF);
  float*  nrm  = (float*)(smem + NRM_OFF);

  const int tid  = threadIdx.x;
  const int lane = tid & 31;
  const int wv   = tid >> 5;           // 8 waves
  const int hl   = lane >> 4;          // half-wave select
  const int ln   = lane & 15;

  const int s  = blockIdx.x >> 1;
  const int i0 = (blockIdx.x & 1) * 128;

  const float* As = A + (size_t)s * Nn * Nn * R1;
  const float* Xs = X + (size_t)s * Nn * CIN;

  // ---- stage x rows (bf16), coalesced ----
  for (int e = tid; e < Nn * CIN; e += 256)
    xrow[e] = f2bf(Xs[e]);

  // ---- pre-pass: A row sums -> norm; also warms this WG's A slice into L2 ----
  for (int q = tid; q < 128 * R; q += 256) {
    const int il = q >> 2, r = q & 3;
    const float* ap = As + ((size_t)(i0 + il) * Nn) * R1 + r;
    __builtin_prefetch(ap, 0, 0);                        // global_prefetch_b8
    float acc = 0.f;
    #pragma unroll 8
    for (int j = 0; j < Nn; ++j) acc += ap[(size_t)j * R1];
    nrm[q] = 1.f / (acc + EPSF);
  }
  __syncthreads();

  const int irow = i0 + wv * 16;   // this wave's output rows
  const int jrow = wv * 32;        // this wave's P-phase rows

  // ---- output accumulators, initialized with bias lin_b[b] ----
  v8f oac[8];
  #pragma unroll
  for (int bt = 0; bt < 8; ++bt) {
    const float b = LB[bt*16 + ln];
    #pragma unroll
    for (int v = 0; v < 8; ++v) oac[bt][v] = b;
  }

  for (int rp = 0; rp < 2; ++rp) {
    // ---- stage W^T for r = 2*rp, 2*rp+1 : wt_[rl][b][a] (contiguous-a rows) ----
    for (int e = tid; e < 2*COUT*CIN; e += 256) {
      const int rl = e >> 14;
      const int b  = (e >> 7) & 127;
      const int a  = e & 127;
      wt_[e] = f2bf(W[((size_t)a * COUT + b) * R + (rp*2 + rl)]);
    }
    __syncthreads();

    // ---- P phase: P_r = x @ W_r ; wave does rows jrow..jrow+31 -> pt_[rl][b][j] ----
    #pragma unroll
    for (int rl = 0; rl < 2; ++rl) {
      const __bf16* wr = wt_ + rl * COUT * CIN;
      #pragma unroll
      for (int jt = 0; jt < 2; ++jt) {
        const int jr = jrow + jt * 16;
        v8f pac[8];
        #pragma unroll
        for (int bt = 0; bt < 8; ++bt)
          #pragma unroll
          for (int v = 0; v < 8; ++v) pac[bt][v] = 0.f;
        #pragma unroll
        for (int kk = 0; kk < 4; ++kk) {             // K = a, 32 per step
          const __bf16* ap = xrow + (jr + ln) * CIN + kk*32 + hl*16;
          const v16bf af = frag(ap, ap + 8);
          #pragma unroll
          for (int bt = 0; bt < 8; ++bt) {
            const __bf16* bp = wr + (bt*16 + ln) * CIN + kk*32 + hl*16;
            pac[bt] = wmma_bf16(af, frag(bp, bp + 8), pac[bt]);
          }
        }
        // store transposed: pt_[rl][b][j]; per lane 8 contiguous bf16 -> one b128 store
        #pragma unroll
        for (int bt = 0; bt < 8; ++bt) {
          v8bf pv;
          #pragma unroll
          for (int v = 0; v < 8; ++v) pv[v] = f2bf(pac[bt][v]);
          *(v8bf*)(pt_ + ((size_t)(rl*COUT + bt*16 + ln)) * Nn + jr + hl*8) = pv;
        }
      }
    }
    __syncthreads();

    // ---- main phase: out += diag(norm_r) * A_r @ P_r  (K = j, 256) ----
    float*  araw = (float*)(smem + ARAW_OFF) + wv * (16*32*2);  // per-wave raw tile
    __bf16* at   = atl + wv * 16 * 32;                          // per-wave bf16 tile
    for (int kk = 0; kk < 8; ++kk) {
      const int j0 = kk * 32;
      // async raw copy: 16 rows x 32 j x 2 r (f32) -> LDS, no VGPR round-trip
      #pragma unroll
      for (int t = 0; t < 32; ++t) {
        const int m  = lane + 32*t;        // 0..1023
        const int p  = m >> 6;             // row in tile
        const int jj = (m >> 1) & 31;      // j within chunk
        const int rr = m & 1;              // r within pair
        const float* g = As + ((size_t)(irow + p) * Nn + (j0 + jj)) * R1 + rp*2 + rr;
        async_copy_b32(g, araw + m);
      }
      wait_async0();                       // ASYNCcnt == 0: tile resident in LDS

      #pragma unroll
      for (int rl = 0; rl < 2; ++rl) {
        const int r = rp*2 + rl;
        const __bf16* pr = pt_ + rl * COUT * Nn;
        // norm-scale + bf16 convert, LDS->LDS (even-bank stride, conflict-free)
        #pragma unroll
        for (int t = 0; t < 16; ++t) {
          const int idx = t*32 + lane;
          at[idx] = f2bf(araw[idx*2 + rl] * nrm[(wv*16 + t)*R + r]);
        }
        const __bf16* afp = at + ln*32 + hl*16;
        const v16bf af = frag(afp, afp + 8);
        #pragma unroll
        for (int bt = 0; bt < 8; ++bt) {
          const __bf16* bp = pr + (bt*16 + ln) * Nn + j0 + hl*16;
          oac[bt] = wmma_bf16(af, frag(bp, bp + 8), oac[bt]);
        }
      }
    }
    __syncthreads();   // protect WT/PT reuse across r-pass
  }

  // ---- linear term: out += x_rows @ lin_w^T ; B[k=a][n=b] = LW[b*CIN + a] (row-major, direct) ----
  #pragma unroll
  for (int kk = 0; kk < 4; ++kk) {
    const __bf16* ap = xrow + (irow + ln) * CIN + kk*32 + hl*16;
    const v16bf af = frag(ap, ap + 8);
    #pragma unroll
    for (int bt = 0; bt < 8; ++bt) {
      const float* bp = LW + (size_t)(bt*16 + ln) * CIN + kk*32 + hl*16;
      const v4f b0 = *(const v4f*)(bp);
      const v4f b1 = *(const v4f*)(bp + 4);
      const v4f b2 = *(const v4f*)(bp + 8);
      const v4f b3 = *(const v4f*)(bp + 12);
      v16bf bfr;
      #pragma unroll
      for (int q = 0; q < 4; ++q) {
        bfr[q]      = f2bf(b0[q]);
        bfr[q + 4]  = f2bf(b1[q]);
        bfr[q + 8]  = f2bf(b2[q]);
        bfr[q + 12] = f2bf(b3[q]);
      }
      oac[bt] = wmma_bf16(af, bfr, oac[bt]);
    }
  }

  // ---- store: C layout -> out[s][irow + v + hl*8][bt*16 + ln] ----
  float* op = OUT + ((size_t)s * Nn + irow) * COUT;
  #pragma unroll
  for (int bt = 0; bt < 8; ++bt)
    #pragma unroll
    for (int v = 0; v < 8; ++v)
      op[(size_t)(v + hl*8) * COUT + bt*16 + ln] = oac[bt][v];
}

} // namespace

extern "C" void kernel_launch(void* const* d_in, const int* in_sizes, int n_in,
                              void* d_out, int out_size, void* d_ws, size_t ws_size,
                              hipStream_t stream) {
  const float* A  = (const float*)d_in[0];   // (128,256,256,5)
  const float* X  = (const float*)d_in[1];   // (128,256,128)
  const float* W  = (const float*)d_in[2];   // (128,128,4)
  const float* LW = (const float*)d_in[3];   // (128,128)
  const float* LB = (const float*)d_in[4];   // (128,)
  float* OUT = (float*)d_out;                // (128,256,128) f32

  dim3 grid(Sn * 2);    // one WG per (s, half of i)
  dim3 block(256);      // 8 wave32
  fused_attnconv<<<grid, block, LDS_BYTES, stream>>>(A, X, W, LW, LB, OUT);
}
